// BigramLanguageModel_4827543241013
// MI455X (gfx1250) — matrix operand order repaired and checked by reading, not compiled, not run on hardware
//
#include <hip/hip_runtime.h>
#include <hip/hip_bf16.h>
#include <math.h>

// ---------------- problem constants ----------------
constexpr int Vv = 65, Dv = 384, Tv = 256, Hv = 6, HSv = 64, Lv = 6, Bv = 128, FFv = 1536;
constexpr int BT = Bv * Tv;          // 32768 tokens
constexpr int NLM = 128;             // LM-head N padded to a multiple of 64

typedef _Float16 v16h __attribute__((ext_vector_type(16)));
typedef _Float16 v8h  __attribute__((ext_vector_type(8)));
typedef float    v8f  __attribute__((ext_vector_type(8)));

// ---------------- WMMA fragment helpers (CDNA5 16x16x32 f16, wave32) ----------------
// A 16x32 (MxK), row-major source: lane l -> row (l&15); holds K = {g*8..g*8+7} U {16+g*8..}
__device__ __forceinline__ v16h load_a_frag(const _Float16* __restrict__ base,
                                            int row, int ld, int kk, int lane) {
    int m = row + (lane & 15);
    int g = lane >> 4;
    const _Float16* p = base + (size_t)m * ld + kk + g * 8;
    v8h lo = *reinterpret_cast<const v8h*>(p);
    v8h hi = *reinterpret_cast<const v8h*>(p + 16);
    v16h r;
#pragma unroll
    for (int i = 0; i < 8; ++i) { r[i] = lo[i]; r[i + 8] = hi[i]; }
    return r;
}
// B 32x16 (KxN), source stored as B^T row-major [N][K]: lane l -> col (l&15),
// holds K = {g*8..g*8+7} U {16+g*8..} (same as A) -> two 16-byte loads.
__device__ __forceinline__ v16h load_b_frag(const _Float16* __restrict__ base,
                                            int col, int ld, int kk, int lane) {
    int n = col + (lane & 15);
    const _Float16* p = base + (size_t)n * ld + kk + (lane >> 4) * 8;
    v8h lo = *reinterpret_cast<const v8h*>(p);
    v8h hi = *reinterpret_cast<const v8h*>(p + 16);
    v16h r;
#pragma unroll
    for (int i = 0; i < 8; ++i) { r[i] = lo[i]; r[i + 8] = hi[i]; }
    return r;
}
__device__ __forceinline__ v8f wmma16(v16h a, v16h b, v8f c) {
    return __builtin_amdgcn_wmma_f32_16x16x32_f16(false, a, false, b, (short)0, c, false, false);
}

// ---------------- embedding ----------------
__global__ void embed_kernel(const int* __restrict__ idx, const float* __restrict__ tok,
                             const float* __restrict__ pos, float* __restrict__ x) {
    int bt = blockIdx.x;
    int token = idx[bt];
    int t = bt & (Tv - 1);
    for (int d = threadIdx.x; d < Dv; d += blockDim.x)
        x[(size_t)bt * Dv + d] = tok[(size_t)token * Dv + d] + pos[(size_t)t * Dv + d];
}

// ---------------- weight convert/transpose to f16 ----------------
// Wq/Wk/Wv: src (H,D,HS) f32 -> dst[(h*HS+e)][d] f16 ([N=D][K=D] B^T layout)
__global__ void cvt_qkv_kernel(const float* __restrict__ src, _Float16* __restrict__ dst, float scale) {
    int i = blockIdx.x * 256 + threadIdx.x;
    if (i >= Hv * Dv * HSv) return;
    int h = i / (Dv * HSv);
    int rem = i - h * Dv * HSv;
    int d = rem / HSv, e = rem - d * HSv;
    dst[(size_t)(h * HSv + e) * Dv + d] = (_Float16)(src[i] * scale);
}
// generic: src [K,N] f32 -> dst [N,K] f16
__global__ void cvt_t_kernel(const float* __restrict__ src, _Float16* __restrict__ dst, int K, int N) {
    int i = blockIdx.x * 256 + threadIdx.x;
    if (i >= K * N) return;
    int k = i / N, n = i - k * N;
    dst[(size_t)n * K + k] = (_Float16)src[i];
}
__global__ void zero_h16_kernel(_Float16* __restrict__ p, int n) {
    int i = blockIdx.x * 256 + threadIdx.x;
    if (i < n) p[i] = (_Float16)0.0f;
}

// ---------------- LayerNorm: f32 in -> f16 out, 1 wave per row ----------------
__global__ __launch_bounds__(256) void ln_kernel(const float* __restrict__ x,
                                                 const float* __restrict__ g,
                                                 const float* __restrict__ bta,
                                                 _Float16* __restrict__ out) {
    int row = blockIdx.x * 8 + (threadIdx.x >> 5);
    int lane = threadIdx.x & 31;
    const float* xr = x + (size_t)row * Dv;
    float s = 0.f, s2 = 0.f;
    for (int i = lane; i < Dv; i += 32) { float t = xr[i]; s += t; s2 += t * t; }
#pragma unroll
    for (int off = 16; off; off >>= 1) {
        s  += __shfl_xor(s,  off, 32);
        s2 += __shfl_xor(s2, off, 32);
    }
    float mean = s * (1.0f / Dv);
    float var  = s2 * (1.0f / Dv) - mean * mean;
    float rstd = rsqrtf(var + 1e-5f);
    _Float16* orow = out + (size_t)row * Dv;
    for (int i = lane; i < Dv; i += 32)
        orow[i] = (_Float16)((xr[i] - mean) * rstd * g[i] + bta[i]);
}

// ---------------- generic WMMA GEMM ----------------
// C[M,N] = A_f16[M,K] @ (WT_f16[N,K])^T  (+bias) (+residual f32) (ReLU?) -> f32 and/or f16
// Grid: (M/128, Npad/64); 8 waves as 4(M) x 2(N); each wave: 32x32 tile.
template <int RELU>
__global__ __launch_bounds__(256) void gemm_f16_kernel(
    const _Float16* __restrict__ A, const _Float16* __restrict__ WT,
    const float* __restrict__ bias, const float* __restrict__ residual,
    float* __restrict__ outF, _Float16* __restrict__ outH,
    int N, int K, int ldc, int Nvalid) {
    int lane = threadIdx.x & 31;
    int wid  = threadIdx.x >> 5;
    int mBase = blockIdx.x * 128 + (wid & 3) * 32;
    int nBase = blockIdx.y * 64  + (wid >> 2) * 32;
    v8f acc[2][2] = {};
    for (int kk = 0; kk < K; kk += 32) {
        v16h a0 = load_a_frag(A, mBase,      K, kk, lane);
        v16h a1 = load_a_frag(A, mBase + 16, K, kk, lane);
        v16h b0 = load_b_frag(WT, nBase,      K, kk, lane);
        v16h b1 = load_b_frag(WT, nBase + 16, K, kk, lane);
        acc[0][0] = wmma16(a0, b0, acc[0][0]);
        acc[0][1] = wmma16(a0, b1, acc[0][1]);
        acc[1][0] = wmma16(a1, b0, acc[1][0]);
        acc[1][1] = wmma16(a1, b1, acc[1][1]);
    }
#pragma unroll
    for (int i = 0; i < 2; ++i)
#pragma unroll
        for (int j = 0; j < 2; ++j) {
            int n = nBase + 16 * j + (lane & 15);
            if (n >= Nvalid) continue;
            int m0 = mBase + 16 * i + 8 * (lane >> 4);
            float bv = bias ? bias[n] : 0.0f;
#pragma unroll
            for (int r = 0; r < 8; ++r) {
                int m = m0 + r;
                float v = acc[i][j][r] + bv;
                if (residual) v += residual[(size_t)m * ldc + n];
                if (RELU) v = fmaxf(v, 0.0f);
                if (outF) outF[(size_t)m * ldc + n] = v;
                if (outH) outH[(size_t)m * N + n] = (_Float16)v;
            }
        }
}

// ---------------- fused causal attention per (b, h, 64-query block) ----------------
// q is pre-scaled by D^-0.5 (folded into Wq conversion).
__global__ __launch_bounds__(256) void attn_kernel(const _Float16* __restrict__ q,
                                                   const _Float16* __restrict__ k,
                                                   const _Float16* __restrict__ v,
                                                   _Float16* __restrict__ o) {
    __shared__ float    sS[64][Tv];     // 64 KB scores
    __shared__ _Float16 sP[64][Tv];     // 32 KB probs
    __shared__ _Float16 sVT[HSv][Tv];   // 32 KB V^T
    int bh = blockIdx.x;
    int b = bh / Hv, h = bh - b * Hv;
    int q0 = blockIdx.y * 64;
    int tid = threadIdx.x, lane = tid & 31, wid = tid >> 5;
    size_t rowBase = (size_t)b * Tv;
    const _Float16* qh = q + rowBase * Dv + h * HSv;
    const _Float16* kh = k + rowBase * Dv + h * HSv;
    const _Float16* vh = v + rowBase * Dv + h * HSv;

    // stage V^T: sVT[e][s] = v[s][e]  (coalesced global reads)
    for (int i = tid; i < HSv * Tv; i += 256) {
        int s = i >> 6, e = i & 63;
        sVT[e][s] = vh[(size_t)s * Dv + e];
    }
    __syncthreads();

    // S = Q K^T : M=64, N=256, K=HS.  8 waves = 2(M) x 4(N), two N passes.
    {
        int wm = wid & 1, wn = wid >> 1;
        for (int half = 0; half < 2; ++half) {
            int nb = half * 128 + wn * 32;
            v8f acc[2][2] = {};
            for (int kk = 0; kk < HSv; kk += 32) {
                v16h a0 = load_a_frag(qh, q0 + wm * 32,      Dv, kk, lane);
                v16h a1 = load_a_frag(qh, q0 + wm * 32 + 16, Dv, kk, lane);
                v16h b0 = load_b_frag(kh, nb,      Dv, kk, lane);  // k rows are B^T layout
                v16h b1 = load_b_frag(kh, nb + 16, Dv, kk, lane);
                acc[0][0] = wmma16(a0, b0, acc[0][0]);
                acc[0][1] = wmma16(a0, b1, acc[0][1]);
                acc[1][0] = wmma16(a1, b0, acc[1][0]);
                acc[1][1] = wmma16(a1, b1, acc[1][1]);
            }
#pragma unroll
            for (int i = 0; i < 2; ++i)
#pragma unroll
                for (int j = 0; j < 2; ++j) {
                    int n = nb + 16 * j + (lane & 15);
#pragma unroll
                    for (int r = 0; r < 8; ++r) {
                        int ml = wm * 32 + 16 * i + 8 * (lane >> 4) + r;
                        sS[ml][n] = (n <= q0 + ml) ? acc[i][j][r] : -1e30f;
                    }
                }
        }
    }
    __syncthreads();

    // softmax: one wave per row, 8 rows per wave
    for (int rr = 0; rr < 8; ++rr) {
        int row = wid * 8 + rr;
        float mx = -1e30f;
        for (int c = lane; c < Tv; c += 32) mx = fmaxf(mx, sS[row][c]);
#pragma unroll
        for (int off = 16; off; off >>= 1) mx = fmaxf(mx, __shfl_xor(mx, off, 32));
        float sum = 0.f;
        for (int c = lane; c < Tv; c += 32) {
            float e = __expf(sS[row][c] - mx);
            sS[row][c] = e;
            sum += e;
        }
#pragma unroll
        for (int off = 16; off; off >>= 1) sum += __shfl_xor(sum, off, 32);
        float inv = 1.0f / sum;
        for (int c = lane; c < Tv; c += 32) sP[row][c] = (_Float16)(sS[row][c] * inv);
    }
    __syncthreads();

    // O = P @ V : M=64, N=HS=64, K=256.  8 waves = 2(M) x 4(N of 16).
    {
        int wm = wid & 1, wn = wid >> 1;
        v8f oc[2] = {};
        for (int kk = 0; kk < Tv; kk += 32) {
            v16h a0 = load_a_frag(&sP[0][0], wm * 32,      Tv, kk, lane);
            v16h a1 = load_a_frag(&sP[0][0], wm * 32 + 16, Tv, kk, lane);
            v16h bf = load_b_frag(&sVT[0][0], wn * 16, Tv, kk, lane);  // sVT is B^T layout
            oc[0] = wmma16(a0, bf, oc[0]);
            oc[1] = wmma16(a1, bf, oc[1]);
        }
        int e = wn * 16 + (lane & 15);
#pragma unroll
        for (int i = 0; i < 2; ++i) {
            int m0 = wm * 32 + 16 * i + 8 * (lane >> 4);
#pragma unroll
            for (int r = 0; r < 8; ++r)
                o[(rowBase + q0 + m0 + r) * Dv + h * HSv + e] = (_Float16)oc[i][r];
        }
    }
}

// ---------------- cross-entropy: per-row loss then deterministic mean ----------------
__global__ __launch_bounds__(256) void rowloss_kernel(const float* __restrict__ logits,
                                                      const int* __restrict__ target,
                                                      float* __restrict__ rl) {
    int row = blockIdx.x * 8 + (threadIdx.x >> 5);
    int lane = threadIdx.x & 31;
    const float* lr = logits + (size_t)row * Vv;
    float mx = -1e30f;
    for (int c = lane; c < Vv; c += 32) mx = fmaxf(mx, lr[c]);
#pragma unroll
    for (int off = 16; off; off >>= 1) mx = fmaxf(mx, __shfl_xor(mx, off, 32));
    float sum = 0.f;
    for (int c = lane; c < Vv; c += 32) sum += __expf(lr[c] - mx);
#pragma unroll
    for (int off = 16; off; off >>= 1) sum += __shfl_xor(sum, off, 32);
    if (lane == 0) rl[row] = (mx + logf(sum)) - lr[target[row]];
}

__global__ void loss_reduce_kernel(const float* __restrict__ rl, float* __restrict__ out) {
    __shared__ float sm[256];
    float s = 0.f;
    for (int i = threadIdx.x; i < BT; i += 256) s += rl[i];
    sm[threadIdx.x] = s;
    __syncthreads();
    for (int off = 128; off; off >>= 1) {
        if ((int)threadIdx.x < off) sm[threadIdx.x] += sm[threadIdx.x + off];
        __syncthreads();
    }
    if (threadIdx.x == 0) out[0] = sm[0] / (float)BT;
}

// ---------------- host orchestration ----------------
extern "C" void kernel_launch(void* const* d_in, const int* in_sizes, int n_in,
                              void* d_out, int out_size, void* d_ws, size_t ws_size,
                              hipStream_t stream) {
    (void)in_sizes; (void)n_in; (void)out_size; (void)ws_size;
    const int*   idx    = (const int*)d_in[0];
    const int*   target = (const int*)d_in[1];
    const float* tok    = (const float*)d_in[2];
    const float* pos    = (const float*)d_in[3];
    const float* ln1g   = (const float*)d_in[4];
    const float* ln1b   = (const float*)d_in[5];
    const float* Wq     = (const float*)d_in[6];
    const float* Wk     = (const float*)d_in[7];
    const float* Wv     = (const float*)d_in[8];
    const float* Wp     = (const float*)d_in[9];
    const float* bp     = (const float*)d_in[10];
    const float* ln2g   = (const float*)d_in[11];
    const float* ln2b   = (const float*)d_in[12];
    const float* W1     = (const float*)d_in[13];
    const float* b1     = (const float*)d_in[14];
    const float* W2     = (const float*)d_in[15];
    const float* b2     = (const float*)d_in[16];
    const float* lnfg   = (const float*)d_in[17];
    const float* lnfb   = (const float*)d_in[18];
    const float* Wlm    = (const float*)d_in[19];
    const float* blm    = (const float*)d_in[20];
    float* out = (float*)d_out;

    char* ws = (char*)d_ws;
    size_t off = 0;
    auto alloc = [&](size_t bytes) -> char* {
        char* p = ws + off;
        off += (bytes + 255) & ~(size_t)255;
        return p;
    };
    float*    x    = (float*)   alloc((size_t)BT * Dv * 4);
    _Float16* h16  = (_Float16*)alloc((size_t)BT * Dv * 2);
    _Float16* q16  = (_Float16*)alloc((size_t)BT * Dv * 2);
    _Float16* k16  = (_Float16*)alloc((size_t)BT * Dv * 2);
    _Float16* v16  = (_Float16*)alloc((size_t)BT * Dv * 2);
    _Float16* o16  = (_Float16*)alloc((size_t)BT * Dv * 2);
    _Float16* ff16 = q16;  // alias: FF=4D, so [q|k|v|o] region == one [BT,FF] buffer;
                           // q/k/v/o are all dead before FFN1 writes ff16.
    _Float16* wqT  = (_Float16*)alloc((size_t)Lv * Dv * Dv * 2);
    _Float16* wkT  = (_Float16*)alloc((size_t)Lv * Dv * Dv * 2);
    _Float16* wvT  = (_Float16*)alloc((size_t)Lv * Dv * Dv * 2);
    _Float16* wpT  = (_Float16*)alloc((size_t)Lv * Dv * Dv * 2);
    _Float16* w1T  = (_Float16*)alloc((size_t)Lv * Dv * FFv * 2);
    _Float16* w2T  = (_Float16*)alloc((size_t)Lv * Dv * FFv * 2);
    _Float16* wlmT = (_Float16*)alloc((size_t)NLM * Dv * 2);
    float*    rls  = (float*)   alloc((size_t)BT * 4);

    // 1. embeddings
    embed_kernel<<<BT, 128, 0, stream>>>(idx, tok, pos, x);

    // 2. weight conversion (f32 -> f16, transposed to [N][K]); softmax scale folded into Wq
    const float scale = 1.0f / sqrtf((float)Dv);
    const int qkvN = Hv * Dv * HSv;
    for (int l = 0; l < Lv; ++l) {
        cvt_qkv_kernel<<<(qkvN + 255) / 256, 256, 0, stream>>>(Wq + (size_t)l * qkvN, wqT + (size_t)l * Dv * Dv, scale);
        cvt_qkv_kernel<<<(qkvN + 255) / 256, 256, 0, stream>>>(Wk + (size_t)l * qkvN, wkT + (size_t)l * Dv * Dv, 1.0f);
        cvt_qkv_kernel<<<(qkvN + 255) / 256, 256, 0, stream>>>(Wv + (size_t)l * qkvN, wvT + (size_t)l * Dv * Dv, 1.0f);
        cvt_t_kernel<<<(Dv * Dv + 255) / 256, 256, 0, stream>>>(Wp + (size_t)l * Dv * Dv, wpT + (size_t)l * Dv * Dv, Dv, Dv);
        cvt_t_kernel<<<(Dv * FFv + 255) / 256, 256, 0, stream>>>(W1 + (size_t)l * Dv * FFv, w1T + (size_t)l * Dv * FFv, Dv, FFv);
        cvt_t_kernel<<<(Dv * FFv + 255) / 256, 256, 0, stream>>>(W2 + (size_t)l * Dv * FFv, w2T + (size_t)l * Dv * FFv, FFv, Dv);
    }
    zero_h16_kernel<<<(NLM * Dv + 255) / 256, 256, 0, stream>>>(wlmT, NLM * Dv);
    cvt_t_kernel<<<(Dv * Vv + 255) / 256, 256, 0, stream>>>(Wlm, wlmT, Dv, Vv);

    // 3. transformer layers
    dim3 g384(BT / 128, Dv / 64);   // (256, 6)
    dim3 gff (BT / 128, FFv / 64);  // (256, 24)
    dim3 glm (BT / 128, NLM / 64);  // (256, 2)
    for (int l = 0; l < Lv; ++l) {
        size_t wdd = (size_t)l * Dv * Dv, wdf = (size_t)l * Dv * FFv;
        ln_kernel<<<BT / 8, 256, 0, stream>>>(x, ln1g + l * Dv, ln1b + l * Dv, h16);
        gemm_f16_kernel<0><<<g384, 256, 0, stream>>>(h16, wqT + wdd, nullptr, nullptr, nullptr, q16, Dv, Dv, Dv, Dv);
        gemm_f16_kernel<0><<<g384, 256, 0, stream>>>(h16, wkT + wdd, nullptr, nullptr, nullptr, k16, Dv, Dv, Dv, Dv);
        gemm_f16_kernel<0><<<g384, 256, 0, stream>>>(h16, wvT + wdd, nullptr, nullptr, nullptr, v16, Dv, Dv, Dv, Dv);
        attn_kernel<<<dim3(Bv * Hv, Tv / 64), 256, 0, stream>>>(q16, k16, v16, o16);
        gemm_f16_kernel<0><<<g384, 256, 0, stream>>>(o16, wpT + wdd, bp + l * Dv, x, x, nullptr, Dv, Dv, Dv, Dv);
        ln_kernel<<<BT / 8, 256, 0, stream>>>(x, ln2g + l * Dv, ln2b + l * Dv, h16);
        gemm_f16_kernel<1><<<gff, 256, 0, stream>>>(h16, w1T + wdf, b1 + (size_t)l * FFv, nullptr, nullptr, ff16, FFv, Dv, FFv, FFv);
        gemm_f16_kernel<0><<<g384, 256, 0, stream>>>(ff16, w2T + wdf, b2 + l * Dv, x, x, nullptr, Dv, FFv, Dv, Dv);
    }

    // 4. final LN + LM head (logits straight into d_out, stride V=65) + loss
    ln_kernel<<<BT / 8, 256, 0, stream>>>(x, lnfg, lnfb, h16);
    gemm_f16_kernel<0><<<glm, 256, 0, stream>>>(h16, wlmT, blm, nullptr, out, nullptr, NLM, Dv, Vv, Vv);
    rowloss_kernel<<<BT / 8, 256, 0, stream>>>(out, target, rls);
    loss_reduce_kernel<<<1, 256, 0, stream>>>(rls, out + (size_t)BT * Vv);
}
